// HyperComputeModule_32615981646424
// MI455X (gfx1250) — matrix operand order repaired and checked by reading, not compiled, not run on hardware
//
#include <hip/hip_runtime.h>

#define BB 8
#define NN 4096
#define DD 256
#define FF 1024

typedef unsigned short ushort_t;
typedef unsigned int uint_t;
typedef __attribute__((ext_vector_type(16))) __bf16 v16bf;
typedef __attribute__((ext_vector_type(8)))  float  v8f;

typedef __attribute__((__vector_size__(4 * sizeof(int)))) int i4_t;
typedef __attribute__((address_space(1))) i4_t gi4_t;   // global int4
typedef __attribute__((address_space(3))) i4_t li4_t;   // LDS int4
typedef __attribute__((address_space(1))) void gvoid;
typedef __attribute__((address_space(3))) void lvoid;

union Frag {
  uint4 q[2];
  v16bf v;
};

__device__ inline ushort_t f2bf(float f) {
  __bf16 h = (__bf16)f;                     // fptrunc, RNE
  return __builtin_bit_cast(unsigned short, h);
}

__device__ inline unsigned ballot32(bool p) {
#if __has_builtin(__builtin_amdgcn_ballot_w32)
  return __builtin_amdgcn_ballot_w32(p);
#else
  return (unsigned)__ballot(p);
#endif
}

// ---- CDNA5 async global->LDS copy (ASYNCcnt path), 16B per lane ----
__device__ inline void async_ld16(const ushort_t* g, ushort_t* l) {
#if __has_builtin(__builtin_amdgcn_global_load_async_to_lds_b128)
  __builtin_amdgcn_global_load_async_to_lds_b128(
      (gi4_t*)(gvoid*)g, (li4_t*)(lvoid*)l, 0, 0);
#else
  unsigned laddr = (unsigned)(size_t)(lvoid*)l;           // 0-based LDS byte offset
  asm volatile("global_load_async_to_lds_b128 %0, %1, off"
               :: "v"(laddr), "v"((unsigned long long)(size_t)g)
               : "memory");
#endif
}

__device__ inline void wait_async0() {
#if __has_builtin(__builtin_amdgcn_s_wait_asynccnt)
  __builtin_amdgcn_s_wait_asynccnt(0);
#else
  asm volatile("s_wait_asynccnt 0x0" ::: "memory");
#endif
}

// stage a full 128x32 bf16 tile (8KB = 512 x 16B chunks, 2 per thread) via async DMA
__device__ inline void stage_bf16_tile(const ushort_t* gbase, int rowStart, int ld, int k0,
                                       ushort_t* lds, int tid) {
  #pragma unroll
  for (int c = 0; c < 2; ++c) {
    int ch  = tid + 256 * c;        // 0..511
    int r   = ch >> 2;              // 0..127
    int off = (ch & 3) * 8;         // ushort offset 0,8,16,24
    async_ld16(&gbase[(size_t)(rowStart + r) * ld + k0 + off], &lds[r * 32 + off]);
  }
}

// stage a 128x32 bf16 tile from a packed bit row: 1 u32 word -> 32 bf16 (0/1)
__device__ inline void stage_bits_tile(const uint_t* gbase, int rowStart, int ldw, int k0,
                                       ushort_t* lds, int tid) {
  if (tid < 128) {
    uint_t w = gbase[(size_t)(rowStart + tid) * ldw + (k0 >> 5)];
    uint_t* dst = (uint_t*)&lds[tid * 32];
    #pragma unroll
    for (int j = 0; j < 16; ++j) {
      dst[j] = (((w >> (2 * j)) & 1u) ? 0x3F80u : 0u)
             | (((w >> (2 * j + 1)) & 1u) ? 0x3F800000u : 0u);
    }
  }
}

// ---------------- epilogue modes ----------------
constexpr int EPI_DIST           = 0; // hg bits from sq-dist threshold
constexpr int EPI_BIAS_T         = 1; // C+bias -> bf16 transposed
constexpr int EPI_SCALE_RES_T    = 2; // C*rowScale + resF32 -> bf16 transposed
constexpr int EPI_SCALE_RES_DUAL = 3; // C*rowScale + resF32 -> f32 + bf16 normal
constexpr int EPI_BIAS_RELU      = 4; // relu(C+bias) -> bf16 normal
constexpr int EPI_BIAS_RES_F32   = 5; // C+bias+resF32 -> f32 normal

// Generic C = A * B^T, A:[M][K], Bop:[Nn][K] (both row-major; A may be packed bits).
// Tile: 128x128x32, 256 threads = 8 waves, each wave 32x64 (2x4 wmma frags).
// Double-buffered LDS, async global->LDS fills, one barrier per K-step.
template<int EPI, bool ABITS>
__global__ __launch_bounds__(256)
void gemm_bf16_wmma(const void* __restrict__ Aop, size_t aBatch, int lda,
                    const ushort_t* __restrict__ Bop, size_t bBatch, int ldb,
                    int M, int Nn, int K,
                    const float* __restrict__ p0, const float* __restrict__ p1,
                    float* __restrict__ fOut, ushort_t* __restrict__ bOut,
                    ushort_t* __restrict__ bOutT, int ldT,
                    ushort_t* __restrict__ bitsOut)
{
  __shared__ ushort_t As[2][128 * 32];
  __shared__ ushort_t Bs[2][128 * 32];
  const int b    = blockIdx.z;
  const int m0   = blockIdx.y * 128;
  const int n0   = blockIdx.x * 128;
  const int tid  = threadIdx.x;
  const int lane = tid & 31;
  const int wave = tid >> 5;
  const int wm0  = (wave >> 1) * 32;   // 4 waves along M
  const int wn0  = (wave & 1) * 64;    // 2 waves along N

  const ushort_t* Ab    = (const ushort_t*)Aop + (size_t)b * aBatch;   // !ABITS
  const uint_t*   AbW   = (const uint_t*)Aop + (size_t)b * aBatch;     // ABITS (lda in words)
  const ushort_t* Bb    = Bop + (size_t)b * bBatch;

  v8f acc[2][4];
  #pragma unroll
  for (int i = 0; i < 2; ++i)
    #pragma unroll
    for (int j = 0; j < 4; ++j)
      #pragma unroll
      for (int v = 0; v < 8; ++v) acc[i][j][v] = 0.0f;

  const int steps = K / 32;
  // prologue: fill buffer 0
  if (ABITS) stage_bits_tile(AbW, m0, lda, 0, As[0], tid);
  else       stage_bf16_tile(Ab, m0, lda, 0, As[0], tid);
  stage_bf16_tile(Bb, n0, ldb, 0, Bs[0], tid);

  for (int i = 0; i < steps; ++i) {
    const int cur = i & 1;
    wait_async0();            // own-wave async fills of buf 'cur' complete
    __syncthreads();          // all waves: fills visible, prior reads of buf 'cur^1' done
    if (i + 1 < steps) {      // prefetch next tile into the other buffer
      const int nk = (i + 1) * 32;
      if (ABITS) stage_bits_tile(AbW, m0, lda, nk, As[cur ^ 1], tid);
      else       stage_bf16_tile(Ab, m0, lda, nk, As[cur ^ 1], tid);
      stage_bf16_tile(Bb, n0, ldb, nk, Bs[cur ^ 1], tid);
    }

    // ---- build fragments per ISA 16x16x32 bf16 layouts ----
    const int c0 = (lane < 16) ? 0 : 8;     // A: lanes<16 K{0-7,16-23}; lanes>=16 K{8-15,24-31}
    const int rb = lane & 15;
    Frag fa[2], fb[4];
    #pragma unroll
    for (int fm = 0; fm < 2; ++fm) {
      int arow = wm0 + fm * 16 + rb;
      fa[fm].q[0] = *(const uint4*)&As[cur][arow * 32 + c0];
      fa[fm].q[1] = *(const uint4*)&As[cur][arow * 32 + 16 + c0];
    }
    const int boff = (lane < 16) ? 0 : 8;   // B: lanes<16 K{0-7,16-23}; lanes>=16 K{8-15,24-31}
    #pragma unroll
    for (int fn = 0; fn < 4; ++fn) {
      int bcol = wn0 + fn * 16 + rb;
      fb[fn].q[0] = *(const uint4*)&Bs[cur][bcol * 32 + boff];
      fb[fn].q[1] = *(const uint4*)&Bs[cur][bcol * 32 + boff + 16];
    }
    #pragma unroll
    for (int fm = 0; fm < 2; ++fm)
      #pragma unroll
      for (int fn = 0; fn < 4; ++fn)
        acc[fm][fn] = __builtin_amdgcn_wmma_f32_16x16x32_bf16(
            false, fa[fm].v, false, fb[fn].v, (short)0, acc[fm][fn], false, false);
  }

  // ---- epilogue. C layout: elem v of lane L = (row base+v+(L>=16?8:0), col base+(L&15)) ----
  const int rhalf = (lane < 16) ? 0 : 8;
  const int ncol  = lane & 15;
  #pragma unroll
  for (int fm = 0; fm < 2; ++fm) {
    #pragma unroll
    for (int fn = 0; fn < 4; ++fn) {
      const int rowTile = m0 + wm0 + fm * 16;       // tile-row origin (lane-independent)
      const int rowBase = rowTile + rhalf;          // per-lane row base (+v)
      const int col     = n0 + wn0 + fn * 16 + ncol;

      if constexpr (EPI == EPI_DIST) {
        const float* n1 = p0 + (size_t)b * M;
        const float* n2 = p1 + (size_t)b * Nn;
        ushort_t* bits = bitsOut + (size_t)b * (size_t)M * (size_t)(Nn >> 4);
        const int jt = (n0 + wn0 + fn * 16) >> 4;
        float n2v = n2[col];
        #pragma unroll
        for (int v = 0; v < 8; ++v) {
          int row = rowBase + v;
          float sq = n1[row] + n2v - 2.0f * acc[fm][fn][v];
          unsigned mask = ballot32(sq < 529.0f);    // dist<23  <=>  sq<23^2
          if (lane == 0)
            bits[(size_t)(rowTile + v) * (Nn >> 4) + jt] = (ushort_t)(mask & 0xffffu);
          else if (lane == 16)
            bits[(size_t)(rowTile + v + 8) * (Nn >> 4) + jt] = (ushort_t)(mask >> 16);
        }
      } else if constexpr (EPI == EPI_BIAS_T) {
        float bias = p0[col];
        alignas(16) ushort_t tmp[8];
        #pragma unroll
        for (int v = 0; v < 8; ++v) tmp[v] = f2bf(acc[fm][fn][v] + bias);
        ushort_t* oT = bOutT + (size_t)b * (size_t)Nn * (size_t)ldT;
        *(uint4*)&oT[(size_t)col * ldT + rowBase] = *(const uint4*)tmp;
      } else if constexpr (EPI == EPI_SCALE_RES_T) {
        const float* sc  = p0 + (size_t)b * M;
        const float* res = p1 + (size_t)b * (size_t)M * Nn;
        alignas(16) ushort_t tmp[8];
        #pragma unroll
        for (int v = 0; v < 8; ++v) {
          int row = rowBase + v;
          tmp[v] = f2bf(acc[fm][fn][v] * sc[row] + res[(size_t)row * Nn + col]);
        }
        ushort_t* oT = bOutT + (size_t)b * (size_t)Nn * (size_t)ldT;
        *(uint4*)&oT[(size_t)col * ldT + rowBase] = *(const uint4*)tmp;
      } else if constexpr (EPI == EPI_SCALE_RES_DUAL) {
        const float* sc  = p0 + (size_t)b * M;
        const float* res = p1 + (size_t)b * (size_t)M * Nn;
        float*    fo = fOut + (size_t)b * (size_t)M * Nn;
        ushort_t* bo = bOut + (size_t)b * (size_t)M * Nn;
        #pragma unroll
        for (int v = 0; v < 8; ++v) {
          int row = rowBase + v;
          float val = acc[fm][fn][v] * sc[row] + res[(size_t)row * Nn + col];
          fo[(size_t)row * Nn + col] = val;
          bo[(size_t)row * Nn + col] = f2bf(val);
        }
      } else if constexpr (EPI == EPI_BIAS_RELU) {
        float bias = p0[col];
        ushort_t* bo = bOut + (size_t)b * (size_t)M * Nn;
        #pragma unroll
        for (int v = 0; v < 8; ++v) {
          int row = rowBase + v;
          bo[(size_t)row * Nn + col] = f2bf(fmaxf(acc[fm][fn][v] + bias, 0.0f));
        }
      } else if constexpr (EPI == EPI_BIAS_RES_F32) {
        float bias = p0[col];
        const float* res = p1 + (size_t)b * (size_t)M * Nn;
        float* fo = fOut + (size_t)b * (size_t)M * Nn;
        #pragma unroll
        for (int v = 0; v < 8; ++v) {
          int row = rowBase + v;
          fo[(size_t)row * Nn + col] = acc[fm][fn][v] + bias + res[(size_t)row * Nn + col];
        }
      }
    }
  }
}

// ---------------- helper kernels ----------------
__global__ void cvt_bf16_kernel(const float* __restrict__ in, ushort_t* __restrict__ out, size_t n) {
  size_t i = (size_t)blockIdx.x * blockDim.x + threadIdx.x;
  if (i < n) out[i] = f2bf(in[i]);
}

__global__ void row_sqnorm_kernel(const float* __restrict__ x, float* __restrict__ out) {
  int row  = blockIdx.x * 8 + (threadIdx.x >> 5);
  int lane = threadIdx.x & 31;
  const float* p = x + (size_t)row * DD;
  float s = 0.0f;
  #pragma unroll
  for (int t = 0; t < DD / 32; ++t) { float v = p[lane + 32 * t]; s += v * v; }
  #pragma unroll
  for (int m = 16; m; m >>= 1) s += __shfl_xor(s, m, 32);
  if (lane == 0) out[row] = s;
}

// transpose the bit matrix (16x16-bit blocks), hgT[j][i] = hg[i][j]
__global__ void bit_transpose_kernel(const ushort_t* __restrict__ in, ushort_t* __restrict__ out) {
  size_t idx = (size_t)blockIdx.x * blockDim.x + threadIdx.x;   // B*256*256 blocks
  int cblk = (int)(idx & 255);
  int rblk = (int)((idx >> 8) & 255);
  int b    = (int)(idx >> 16);
  const ushort_t* ib = in  + (size_t)b * NN * (NN >> 4);
  ushort_t*       ob = out + (size_t)b * NN * (NN >> 4);
  ushort_t r[16];
  #pragma unroll
  for (int i = 0; i < 16; ++i) r[i] = ib[(size_t)(rblk * 16 + i) * (NN >> 4) + cblk];
  #pragma unroll
  for (int c = 0; c < 16; ++c) {
    unsigned t = 0;
    #pragma unroll
    for (int i = 0; i < 16; ++i) t |= ((unsigned)(r[i] >> c) & 1u) << i;
    ob[(size_t)(cblk * 16 + c) * (NN >> 4) + rblk] = (ushort_t)t;
  }
}

// norm[row] = 1/popcount(row bits) or 0
__global__ void deg_norm_kernel(const ushort_t* __restrict__ bits, float* __restrict__ norm) {
  int row  = blockIdx.x * 8 + (threadIdx.x >> 5);
  int lane = threadIdx.x & 31;
  const uint_t* w = (const uint_t*)(bits + (size_t)row * (NN >> 4));   // 128 words
  int cnt = 0;
  #pragma unroll
  for (int t = 0; t < 4; ++t) cnt += __popc(w[lane + 32 * t]);
  #pragma unroll
  for (int m = 16; m; m >>= 1) cnt += __shfl_xor(cnt, m, 32);
  if (lane == 0) norm[row] = cnt ? 1.0f / (float)cnt : 0.0f;
}

__global__ void layernorm_kernel(float* __restrict__ q,
                                 const float* __restrict__ g, const float* __restrict__ bta) {
  int row  = blockIdx.x * 8 + (threadIdx.x >> 5);
  int lane = threadIdx.x & 31;
  float* p = q + (size_t)row * DD;
  float vals[DD / 32];
  float s = 0.0f, ss = 0.0f;
  #pragma unroll
  for (int t = 0; t < DD / 32; ++t) {
    float v = p[lane + 32 * t];
    vals[t] = v; s += v; ss += v * v;
  }
  #pragma unroll
  for (int m = 16; m; m >>= 1) { s += __shfl_xor(s, m, 32); ss += __shfl_xor(ss, m, 32); }
  float mu  = s * (1.0f / DD);
  float var = ss * (1.0f / DD) - mu * mu;
  float rs  = rsqrtf(var + 1e-5f);
  #pragma unroll
  for (int t = 0; t < DD / 32; ++t) {
    int c = lane + 32 * t;
    p[c] = (vals[t] - mu) * rs * g[c] + bta[c];
  }
}

// ---------------- launcher ----------------
extern "C" void kernel_launch(void* const* d_in, const int* in_sizes, int n_in,
                              void* d_out, int out_size, void* d_ws, size_t ws_size,
                              hipStream_t stream) {
  (void)in_sizes; (void)n_in; (void)out_size; (void)ws_size;
  const float* f2_disc = (const float*)d_in[0];  // discriminative_query  [B,N,D]
  const float* f1_ind  = (const float*)d_in[1];  // indiscriminative_query[B,N,D]
  const float* W_fc    = (const float*)d_in[2];  // [D,D]
  const float* b_fc    = (const float*)d_in[3];  // [D]
  const float* W1      = (const float*)d_in[4];  // [F,D]
  const float* b1      = (const float*)d_in[5];  // [F]
  const float* W2      = (const float*)d_in[6];  // [D,F]
  const float* b2      = (const float*)d_in[7];  // [D]
  const float* ln_g    = (const float*)d_in[8];  // [D]
  const float* ln_b    = (const float*)d_in[9];  // [D]
  float* out = (float*)d_out;                    // q -> LN in place

  char* ws = (char*)d_ws;
  size_t off = 0;
  auto alloc = [&](size_t bytes) -> void* {
    void* p = ws + off;
    off += (bytes + 255) & ~(size_t)255;
    return p;
  };
  const size_t BND = (size_t)BB * NN * DD;
  ushort_t* f1_bf   = (ushort_t*)alloc(BND * 2);
  ushort_t* f2_bf   = (ushort_t*)alloc(BND * 2);
  ushort_t* Wfc_bf  = (ushort_t*)alloc((size_t)DD * DD * 2);
  ushort_t* W1_bf   = (ushort_t*)alloc((size_t)FF * DD * 2);
  ushort_t* W2_bf   = (ushort_t*)alloc((size_t)DD * FF * 2);
  float*    n1      = (float*)alloc((size_t)BB * NN * 4);
  float*    n2      = (float*)alloc((size_t)BB * NN * 4);
  ushort_t* hg16    = (ushort_t*)alloc((size_t)BB * NN * (NN / 16) * 2);
  ushort_t* hgT16   = (ushort_t*)alloc((size_t)BB * NN * (NN / 16) * 2);
  float*    norm_v  = (float*)alloc((size_t)BB * NN * 4);
  float*    norm_e  = (float*)alloc((size_t)BB * NN * 4);
  ushort_t* Xt      = (ushort_t*)alloc(BND * 2);           // [B][D][N]
  ushort_t* Et      = (ushort_t*)alloc(BND * 2);           // [B][D][N]
  ushort_t* enh_bf  = (ushort_t*)alloc(BND * 2);           // [B][N][D]
  float*    enh_f32 = (float*)alloc(BND * 4);              // [B][N][D]
  ushort_t* h_bf    = (ushort_t*)alloc((size_t)BB * NN * FF * 2); // [B][N][F]

  dim3 blk(256);

  // bf16 conversions
  cvt_bf16_kernel<<<(unsigned)(BND / 256), blk, 0, stream>>>(f1_ind,  f1_bf,  BND);
  cvt_bf16_kernel<<<(unsigned)(BND / 256), blk, 0, stream>>>(f2_disc, f2_bf,  BND);
  cvt_bf16_kernel<<<(DD * DD) / 256, blk, 0, stream>>>(W_fc, Wfc_bf, (size_t)DD * DD);
  cvt_bf16_kernel<<<(FF * DD) / 256, blk, 0, stream>>>(W1,   W1_bf,  (size_t)FF * DD);
  cvt_bf16_kernel<<<(DD * FF) / 256, blk, 0, stream>>>(W2,   W2_bf,  (size_t)DD * FF);

  // squared row norms (f32 inputs for accuracy)
  row_sqnorm_kernel<<<(BB * NN) / 8, blk, 0, stream>>>(f1_ind,  n1);
  row_sqnorm_kernel<<<(BB * NN) / 8, blk, 0, stream>>>(f2_disc, n2);

  // G1: S = f1·f2^T ; hg bits = (||f1||^2+||f2||^2-2S < 23^2)
  gemm_bf16_wmma<EPI_DIST, false><<<dim3(NN / 128, NN / 128, BB), blk, 0, stream>>>(
      f1_bf, BND / BB, DD, f2_bf, BND / BB, DD, NN, NN, DD,
      n1, n2, nullptr, nullptr, nullptr, 0, hg16);

  // hgT bits + degree norms
  bit_transpose_kernel<<<(BB * 256 * 256) / 256, blk, 0, stream>>>(hg16, hgT16);
  deg_norm_kernel<<<(BB * NN) / 8, blk, 0, stream>>>(hg16,  norm_v);
  deg_norm_kernel<<<(BB * NN) / 8, blk, 0, stream>>>(hgT16, norm_e);

  // G2: X = f1·W_fc^T + b_fc, stored transposed -> Xt [B][D][N]
  gemm_bf16_wmma<EPI_BIAS_T, false><<<dim3(DD / 128, NN / 128, BB), blk, 0, stream>>>(
      f1_bf, BND / BB, DD, Wfc_bf, 0, DD, NN, DD, DD,
      b_fc, nullptr, nullptr, nullptr, Xt, NN, nullptr);

  // G3: E = norm_e ⊙ (hgT·X) + f2, stored transposed -> Et [B][D][N]
  gemm_bf16_wmma<EPI_SCALE_RES_T, true><<<dim3(DD / 128, NN / 128, BB), blk, 0, stream>>>(
      hgT16, (size_t)NN * (NN / 32), NN / 32, Xt, (size_t)DD * NN, NN, NN, DD, NN,
      norm_e, f2_disc, nullptr, nullptr, Et, NN, nullptr);

  // G4: enhanced = norm_v ⊙ (hg·E) + f1 -> enh_f32 + enh_bf [B][N][D]
  gemm_bf16_wmma<EPI_SCALE_RES_DUAL, true><<<dim3(DD / 128, NN / 128, BB), blk, 0, stream>>>(
      hg16, (size_t)NN * (NN / 32), NN / 32, Et, (size_t)DD * NN, NN, NN, DD, NN,
      norm_v, f1_ind, enh_f32, enh_bf, nullptr, 0, nullptr);

  // G5: h = relu(enhanced·W1^T + b1) -> h_bf [B][N][F]
  gemm_bf16_wmma<EPI_BIAS_RELU, false><<<dim3(FF / 128, NN / 128, BB), blk, 0, stream>>>(
      enh_bf, BND / BB, DD, W1_bf, 0, DD, NN, FF, DD,
      b1, nullptr, nullptr, h_bf, nullptr, 0, nullptr);

  // G6: q = enhanced + h·W2^T + b2 -> d_out (f32)
  gemm_bf16_wmma<EPI_BIAS_RES_F32, false><<<dim3(DD / 128, NN / 128, BB), blk, 0, stream>>>(
      h_bf, (size_t)NN * FF, FF, W2_bf, 0, FF, NN, DD, FF,
      b2, enh_f32, out, nullptr, nullptr, 0, nullptr);

  // LayerNorm in place on d_out
  layernorm_kernel<<<(BB * NN) / 8, blk, 0, stream>>>(out, ln_g, ln_b);
}
